// Decoder_79714593013989
// MI455X (gfx1250) — matrix operand order repaired and checked by reading, not compiled, not run on hardware
//
#include <hip/hip_runtime.h>
#include <hip/hip_bf16.h>
#include <stdint.h>

// ---------------------------------------------------------------------------
// Dims (fixed by the reference)
// ---------------------------------------------------------------------------
#define TT   64
#define BB   32
#define VV   32000
#define HH   1024
#define EMB  512
#define K0   (EMB + HH + HH)   // 2560 : [emb | feed | h0_prev]
#define K1   (HH + HH)         // 2048 : [h0_new | h1_prev]
#define EPSF 1e-7f

typedef __bf16 bf16_t;
typedef __attribute__((ext_vector_type(16))) __bf16 v16bf;
typedef __attribute__((ext_vector_type(8)))  float  v8f;

// ---------------------------------------------------------------------------
// Helpers
// ---------------------------------------------------------------------------
__device__ __forceinline__ unsigned short f2bf(float f) {
    unsigned int u = __float_as_uint(f);
    u = (u + 0x7FFFu + ((u >> 16) & 1u)) >> 16;
    return (unsigned short)u;
}

__device__ __forceinline__ float sigmoidf_(float x) {
    return 1.0f / (1.0f + __expf(-x));
}

union V16U { v16bf v; uint4 q[2]; };

// A fragment: two b128 loads at +k and +k+16 elements from a per-lane base.
__device__ __forceinline__ v16bf frag_at(const unsigned short* rowp, int k) {
    V16U u;
    u.q[0] = *reinterpret_cast<const uint4*>(rowp + k);
    u.q[1] = *reinterpret_cast<const uint4*>(rowp + k + 16);
    return u.v;
}

// B fragment: two b128 loads at +k and +k+16 elements from a per-lane base.
__device__ __forceinline__ v16bf bfrag_at(const unsigned short* rowp, int k) {
    V16U u;
    u.q[0] = *reinterpret_cast<const uint4*>(rowp + k);
    u.q[1] = *reinterpret_cast<const uint4*>(rowp + k + 16);
    return u.v;
}

__device__ __forceinline__ v8f wmma_bf16(v16bf a, v16bf b, v8f c) {
    return __builtin_amdgcn_wmma_f32_16x16x32_bf16(
        false, a, false, b, (short)0, c, false, false);
}

// ---------------------------------------------------------------------------
// Software-pipelined 4-adjacent-tile WMMA core with COMPILE-TIME strides:
// one A base + one W base per lane; the 2nd A row and B tiles 1..3 are at
// constant element offsets (16*LDA, i*16*KDIM <= 240KB) that fold into the
// 24-bit instruction offsets -> 2 pointer pairs total, no 64-bit address
// adds and no spills in the loop.  B fragments double-buffered so the next
// K-step's 8 weight loads are in flight under the current 8 WMMAs.
// acc pairing: acc[2i] (M rows 0-15) / acc[2i+1] (M rows 16-31).
// ---------------------------------------------------------------------------
template<int KDIM, int LDA>
__device__ __forceinline__ void mma_core(const unsigned short* abase,
                                         const unsigned short* wbase,
                                         v8f* acc) {
    constexpr int TS = 16 * KDIM;   // B tile stride (elements)
    constexpr int AS = 16 * LDA;    // A M-half stride (elements)
    v16bf b0 = bfrag_at(wbase, 0 * TS);
    v16bf b1 = bfrag_at(wbase, 1 * TS);
    v16bf b2 = bfrag_at(wbase, 2 * TS);
    v16bf b3 = bfrag_at(wbase, 3 * TS);
    for (int k = 0; k + 32 < KDIM; k += 32) {
        int kn = k + 32;
        v16bf a0 = frag_at(abase, k);
        v16bf a1 = frag_at(abase, k + AS);
        v16bf nb0 = bfrag_at(wbase, 0 * TS + kn);
        v16bf nb1 = bfrag_at(wbase, 1 * TS + kn);
        v16bf nb2 = bfrag_at(wbase, 2 * TS + kn);
        v16bf nb3 = bfrag_at(wbase, 3 * TS + kn);
        acc[0] = wmma_bf16(a0, b0, acc[0]);
        acc[1] = wmma_bf16(a1, b0, acc[1]);
        acc[2] = wmma_bf16(a0, b1, acc[2]);
        acc[3] = wmma_bf16(a1, b1, acc[3]);
        acc[4] = wmma_bf16(a0, b2, acc[4]);
        acc[5] = wmma_bf16(a1, b2, acc[5]);
        acc[6] = wmma_bf16(a0, b3, acc[6]);
        acc[7] = wmma_bf16(a1, b3, acc[7]);
        b0 = nb0; b1 = nb1; b2 = nb2; b3 = nb3;
    }
    {
        constexpr int kl = KDIM - 32;
        v16bf a0 = frag_at(abase, kl);
        v16bf a1 = frag_at(abase, kl + AS);
        acc[0] = wmma_bf16(a0, b0, acc[0]);
        acc[1] = wmma_bf16(a1, b0, acc[1]);
        acc[2] = wmma_bf16(a0, b1, acc[2]);
        acc[3] = wmma_bf16(a1, b1, acc[3]);
        acc[4] = wmma_bf16(a0, b2, acc[4]);
        acc[5] = wmma_bf16(a1, b2, acc[5]);
        acc[6] = wmma_bf16(a0, b3, acc[6]);
        acc[7] = wmma_bf16(a1, b3, acc[7]);
    }
}

// ---------------------------------------------------------------------------
// One-time setup kernels
// ---------------------------------------------------------------------------
__global__ void convert_bf16_kernel(unsigned short* dst, int dld, int doff,
                                    const float* src, int sw, long total) {
    long stride = (long)gridDim.x * blockDim.x;
    for (long i = (long)blockIdx.x * blockDim.x + threadIdx.x; i < total; i += stride) {
        long r = i / sw;
        int  c = (int)(i - r * sw);
        dst[r * dld + doff + c] = f2bf(src[i]);
    }
}

// Gate-interleaved conversion for fused LSTM weights: source row
// n = g*HH + j*16 + r  ->  dest row  j*64 + g*16 + r, so that one wave's
// four gate tiles for hidden-column block j are ADJACENT 16-row tiles
// (constant 16*KDIM offsets in the MMA core instead of HH*KDIM).
__global__ void convert_bf16_gates_kernel(unsigned short* dst, int dld, int doff,
                                          const float* src, int sw, long total) {
    long stride = (long)gridDim.x * blockDim.x;
    for (long i = (long)blockIdx.x * blockDim.x + threadIdx.x; i < total; i += stride) {
        long n = i / sw;                 // source row in [0, 4H)
        int  c = (int)(i - n * sw);      // source col
        int  g = (int)(n >> 10);         // gate  (n / HH)
        int  h = (int)(n & (HH - 1));    // hidden col (n % HH)
        int  j = h >> 4;
        int  r = h & 15;
        long drow = (long)j * 64 + g * 16 + r;
        dst[drow * dld + doff + c] = f2bf(src[i]);
    }
}

__global__ void zero_f32_kernel(float* p, int n) {
    int i = blockIdx.x * blockDim.x + threadIdx.x;
    if (i < n) p[i] = 0.0f;
}

__global__ void copy_f32_kernel(float* dst, const float* src, int n) {
    int i = blockIdx.x * blockDim.x + threadIdx.x;
    if (i < n) dst[i] = src[i];
}

__global__ void addvec_kernel(float* dst, const float* a, const float* b, int n) {
    int i = blockIdx.x * blockDim.x + threadIdx.x;
    if (i < n) dst[i] = a[i] + b[i];
}

// ---------------------------------------------------------------------------
// Per-step: stage bf16 A matrices. A0 = [emb | feed | h0_prev] (32 x 2560),
// A1 tail = h1_prev into A1[:, 1024:2048] (A1 head written by LSTM0 epilogue).
// ---------------------------------------------------------------------------
__global__ void prep_inputs_kernel(const int* tokens, int t, const float* E,
                                   const float* feed, const float* h0s,
                                   const float* h1s,
                                   unsigned short* A0, unsigned short* A1) {
    int idx = blockIdx.x * blockDim.x + threadIdx.x;
    const int nA0 = BB * K0;                     // 81920
    if (idx < nA0) {
        int m = idx / K0;
        int c = idx - m * K0;
        float v;
        if (c < EMB) {
            int tok = tokens[t * BB + m];
            v = E[(size_t)tok * EMB + c];
        } else if (c < EMB + HH) {
            v = feed[m * HH + (c - EMB)];
        } else {
            v = h0s[m * HH + (c - EMB - HH)];
        }
        A0[idx] = f2bf(v);
    } else {
        int j = idx - nA0;
        if (j < BB * HH) {
            int m = j / HH;
            int c = j - m * HH;
            A1[m * K1 + HH + c] = f2bf(h1s[m * HH + c]);
        }
    }
}

// ---------------------------------------------------------------------------
// Fused LSTM cell: g = A @ Wcat.T + bsum, gates -> (h, c).
// Weights are stored gate-interleaved (see convert_bf16_gates_kernel), so
// one wave's four gate tiles for its 16 hidden columns are the 4 ADJACENT
// tiles starting at row wave*64 -> same mma_core as the GEMMs, and the
// i/f/g/o values for a given (m, col) land in the same lane/element of the
// four accumulators -> register-local sigmoid/tanh epilogue.
// Launch: H/16 = 64 waves -> <<<16, 128>>>.
// ---------------------------------------------------------------------------
template<int KDIM>
__global__ void lstm_gate_kernel(const unsigned short* A,
                                 const unsigned short* Wcat, const float* bsum,
                                 float* c_state, float* h_out, float* h_out2,
                                 unsigned short* hbf, int hbf_ld) {
    int wave = (blockIdx.x * blockDim.x + threadIdx.x) >> 5;
    int lane = threadIdx.x & 31;
    int col0 = wave * 16;
    if (col0 >= HH) return;

    int lm = lane & 15;
    int kh = lane >> 4;
    const unsigned short* abase = A + (size_t)lm * KDIM + 8 * kh;
    const unsigned short* wbase = Wcat + (size_t)(wave * 64 + lm) * KDIM + 8 * kh;

    v8f acc[8];   // acc[g*2 + mhalf]  (gate g = tile g = acc[2g], acc[2g+1])
    #pragma unroll
    for (int i = 0; i < 8; ++i) acc[i] = v8f{};

    mma_core<KDIM, KDIM>(abase, wbase, acc);

    int n  = col0 + lm;
    int mb = kh * 8;
    float bi_ = bsum[0 * HH + n];
    float bf_ = bsum[1 * HH + n];
    float bg_ = bsum[2 * HH + n];
    float bo_ = bsum[3 * HH + n];

    #pragma unroll
    for (int mh = 0; mh < 2; ++mh) {
        #pragma unroll
        for (int r = 0; r < 8; ++r) {
            int m = mh * 16 + mb + r;
            float gi = acc[0 + mh][r] + bi_;
            float gf = acc[2 + mh][r] + bf_;
            float gg = acc[4 + mh][r] + bg_;
            float go = acc[6 + mh][r] + bo_;
            float cc = sigmoidf_(gf) * c_state[m * HH + n] + sigmoidf_(gi) * tanhf(gg);
            float hh = sigmoidf_(go) * tanhf(cc);
            c_state[m * HH + n] = cc;
            h_out[m * HH + n] = hh;
            if (h_out2) h_out2[m * HH + n] = hh;
            hbf[m * hbf_ld + n] = f2bf(hh);
        }
    }
}

// ---------------------------------------------------------------------------
// Skinny GEMM, 4 adjacent N-tiles per wave: C[32,N] = A[32,K] @ W[N,K].T + b
// Launch: N/64 waves.  Optional bf16 mirror of C.
// ---------------------------------------------------------------------------
template<int KDIM, int LDA>
__global__ void gemm32x4_kernel(const unsigned short* A,
                                const unsigned short* W, const float* bias,
                                float* Cout, int ldc,
                                unsigned short* Cbf, int ldcbf, int N) {
    int wave = (blockIdx.x * blockDim.x + threadIdx.x) >> 5;
    int lane = threadIdx.x & 31;
    int n0 = wave * 64;
    if (n0 >= N) return;

    int lm = lane & 15;
    int kh = lane >> 4;
    const unsigned short* abase = A + (size_t)lm * LDA + 8 * kh;
    const unsigned short* wbase = W + (size_t)(n0 + lm) * KDIM + 8 * kh;

    v8f acc[8];   // acc[tile*2 + mhalf]
    #pragma unroll
    for (int i = 0; i < 8; ++i) acc[i] = v8f{};

    mma_core<KDIM, LDA>(abase, wbase, acc);

    int mb = kh * 8;
    #pragma unroll
    for (int i = 0; i < 4; ++i) {
        int n = n0 + 16 * i + lm;
        float bi = bias ? bias[n] : 0.0f;
        #pragma unroll
        for (int r = 0; r < 8; ++r) {
            float v0 = acc[i * 2 + 0][r] + bi;
            float v1 = acc[i * 2 + 1][r] + bi;
            Cout[(size_t)(mb + r) * ldc + n]      = v0;
            Cout[(size_t)(16 + mb + r) * ldc + n] = v1;
            if (Cbf) {
                Cbf[(size_t)(mb + r) * ldcbf + n]      = f2bf(v0);
                Cbf[(size_t)(16 + mb + r) * ldcbf + n] = f2bf(v1);
            }
        }
    }
}

// ---------------------------------------------------------------------------
// Attention for step t: scores over t'<=t, softmax, scatter into copybuf,
// summary -> bf16 into A2[:, 1024:2048].  grid = B, block = 256 (8 waves).
// ---------------------------------------------------------------------------
__global__ void attention_kernel(const float* keyv, const float* hbuf,
                                 const float* htop, const int* tokens, int t,
                                 float* copybuf, unsigned short* A2) {
    __shared__ float sc[TT];
    int b    = blockIdx.x;
    int tid  = threadIdx.x;
    int wave = tid >> 5;
    int lane = tid & 31;

    for (int tp = wave; tp <= t; tp += 8) {
        const float* kv = keyv + ((size_t)tp * BB + b) * HH;
        const float* ht = htop + (size_t)b * HH;
        float s = 0.0f;
        for (int k = lane; k < HH; k += 32) s += kv[k] * ht[k];
        #pragma unroll
        for (int o = 16; o > 0; o >>= 1) s += __shfl_down(s, o, 32);
        if (lane == 0) {
            if (tokens[tp * BB + b] == 0) s += -99999.0f;   // PAD mask
            sc[tp] = s;
        }
    }
    __syncthreads();

    if (tid == 0) {
        float mx = -1e30f;
        for (int tp = 0; tp <= t; ++tp) mx = fmaxf(mx, sc[tp]);
        float sm = 0.0f;
        for (int tp = 0; tp <= t; ++tp) { float e = __expf(sc[tp] - mx); sc[tp] = e; sm += e; }
        float inv = 1.0f / sm;
        for (int tp = 0; tp <= t; ++tp) sc[tp] *= inv;
    }
    __syncthreads();

    if (tid <= t) {
        atomicAdd(&copybuf[(size_t)b * VV + tokens[tid * BB + b]], sc[tid]);
    }

    for (int k = tid; k < HH; k += 256) {
        float s = 0.0f;
        for (int tp = 0; tp <= t; ++tp)
            s += hbuf[((size_t)tp * BB + b) * HH + k] * sc[tp];
        A2[(size_t)b * K1 + HH + k] = f2bf(s);
    }
}

// ---------------------------------------------------------------------------
// Row softmax stats over V=32000 and copy-weight cw = p[:, COPY_ID=3].
// ---------------------------------------------------------------------------
__global__ void rowstats_kernel(const float* logits, float* maxv, float* sumv,
                                float* cwv) {
    __shared__ float red[256];
    int b = blockIdx.x;
    int tid = threadIdx.x;
    const float* row = logits + (size_t)b * VV;

    float mx = -1e30f;
    for (int v = tid; v < VV; v += 256) mx = fmaxf(mx, row[v]);
    red[tid] = mx; __syncthreads();
    for (int o = 128; o > 0; o >>= 1) {
        if (tid < o) red[tid] = fmaxf(red[tid], red[tid + o]);
        __syncthreads();
    }
    mx = red[0]; __syncthreads();

    float sm = 0.0f;
    for (int v = tid; v < VV; v += 256) sm += __expf(row[v] - mx);
    red[tid] = sm; __syncthreads();
    for (int o = 128; o > 0; o >>= 1) {
        if (tid < o) red[tid] += red[tid + o];
        __syncthreads();
    }
    if (tid == 0) {
        maxv[b] = mx;
        sumv[b] = red[0];
        cwv[b]  = __expf(row[3] - mx) / red[0];
    }
}

// out = log(cw*(copy+eps) + (1-cw)*p)
__global__ void output_kernel(const float* logits, const float* maxv,
                              const float* sumv, const float* cwv,
                              const float* copybuf, float* out, int t) {
    int idx = blockIdx.x * blockDim.x + threadIdx.x;
    if (idx >= BB * VV) return;
    int b = idx / VV;
    int v = idx - b * VV;
    float p  = __expf(logits[idx] - maxv[b]) / sumv[b];
    float cp = copybuf[idx] + EPSF;
    float cw = cwv[b];
    out[((size_t)t * BB + b) * VV + v] = logf(cw * cp + (1.0f - cw) * p);
}

// Re-zero only the copybuf entries touched at this step (<< full memset).
__global__ void copyclear_kernel(const int* tokens, float* copybuf) {
    int tp = blockIdx.x;
    int b  = threadIdx.x;
    copybuf[(size_t)b * VV + tokens[tp * BB + b]] = 0.0f;
}

// ---------------------------------------------------------------------------
// Host orchestration
// ---------------------------------------------------------------------------
extern "C" void kernel_launch(void* const* d_in, const int* in_sizes, int n_in,
                              void* d_out, int out_size, void* d_ws, size_t ws_size,
                              hipStream_t stream) {
    (void)in_sizes; (void)n_in; (void)out_size; (void)ws_size;

    const int*   tokens = (const int*)  d_in[0];
    const float* h0in   = (const float*)d_in[1];
    const float* c0in   = (const float*)d_in[2];
    const float* E      = (const float*)d_in[3];
    const float* Wih0   = (const float*)d_in[4];
    const float* Whh0   = (const float*)d_in[5];
    const float* bih0   = (const float*)d_in[6];
    const float* bhh0   = (const float*)d_in[7];
    const float* Wih1   = (const float*)d_in[8];
    const float* Whh1   = (const float*)d_in[9];
    const float* bih1   = (const float*)d_in[10];
    const float* bhh1   = (const float*)d_in[11];
    const float* Wk     = (const float*)d_in[12];
    const float* bk     = (const float*)d_in[13];
    const float* Wc     = (const float*)d_in[14];
    const float* bc     = (const float*)d_in[15];
    const float* Wp     = (const float*)d_in[16];
    const float* bp     = (const float*)d_in[17];
    float* out          = (float*)d_out;

    // ---- workspace layout (256B aligned slices) ----
    char* ws = (char*)d_ws;
    size_t off = 0;
    auto alloc = [&](size_t bytes) -> void* {
        off = (off + 255) & ~(size_t)255;
        void* p = ws + off;
        off += bytes;
        return p;
    };
    unsigned short* Wp_bf  = (unsigned short*)alloc((size_t)VV * HH * 2);
    unsigned short* W0_bf  = (unsigned short*)alloc((size_t)4 * HH * K0 * 2);
    unsigned short* W1_bf  = (unsigned short*)alloc((size_t)4 * HH * K1 * 2);
    unsigned short* Wk_bf  = (unsigned short*)alloc((size_t)HH * HH * 2);
    unsigned short* Wc_bf  = (unsigned short*)alloc((size_t)HH * K1 * 2);
    float* bsum0           = (float*)alloc(4 * HH * 4);
    float* bsum1           = (float*)alloc(4 * HH * 4);
    unsigned short* A0     = (unsigned short*)alloc((size_t)BB * K0 * 2);
    unsigned short* A1     = (unsigned short*)alloc((size_t)BB * K1 * 2);
    unsigned short* A2     = (unsigned short*)alloc((size_t)BB * K1 * 2);
    unsigned short* combbf = (unsigned short*)alloc((size_t)BB * HH * 2);
    float* feed            = (float*)alloc((size_t)BB * HH * 4);
    float* h0s             = (float*)alloc((size_t)BB * HH * 4);
    float* c0s             = (float*)alloc((size_t)BB * HH * 4);
    float* h1s             = (float*)alloc((size_t)BB * HH * 4);
    float* c1s             = (float*)alloc((size_t)BB * HH * 4);
    float* hbuf            = (float*)alloc((size_t)TT * BB * HH * 4);
    float* keyv            = (float*)alloc((size_t)TT * BB * HH * 4);
    float* copybuf         = (float*)alloc((size_t)BB * VV * 4);
    float* logits          = (float*)alloc((size_t)BB * VV * 4);
    float* maxv            = (float*)alloc(BB * 4);
    float* sumv            = (float*)alloc(BB * 4);
    float* cwv             = (float*)alloc(BB * 4);

    // ---- one-time: bf16 weight conversion (stays resident in 192MB L2) ----
    // LSTM fused weights are written gate-interleaved (adjacent gate tiles).
    const int CT = 256, CG = 4096;
    convert_bf16_gates_kernel<<<CG, CT, 0, stream>>>(W0_bf, K0, 0,        Wih0, EMB + HH, (long)4 * HH * (EMB + HH));
    convert_bf16_gates_kernel<<<CG, CT, 0, stream>>>(W0_bf, K0, EMB + HH, Whh0, HH,       (long)4 * HH * HH);
    convert_bf16_gates_kernel<<<CG, CT, 0, stream>>>(W1_bf, K1, 0,        Wih1, HH,       (long)4 * HH * HH);
    convert_bf16_gates_kernel<<<CG, CT, 0, stream>>>(W1_bf, K1, HH,       Whh1, HH,       (long)4 * HH * HH);
    convert_bf16_kernel<<<CG, CT, 0, stream>>>(Wk_bf, HH, 0, Wk, HH, (long)HH * HH);
    convert_bf16_kernel<<<CG, CT, 0, stream>>>(Wc_bf, K1, 0, Wc, K1, (long)HH * K1);
    convert_bf16_kernel<<<CG, CT, 0, stream>>>(Wp_bf, HH, 0, Wp, HH, (long)VV * HH);

    addvec_kernel<<<(4 * HH + 255) / 256, 256, 0, stream>>>(bsum0, bih0, bhh0, 4 * HH);
    addvec_kernel<<<(4 * HH + 255) / 256, 256, 0, stream>>>(bsum1, bih1, bhh1, 4 * HH);

    // ---- per-launch state init (deterministic across graph replays) ----
    zero_f32_kernel<<<(BB * HH + 255) / 256, 256, 0, stream>>>(feed, BB * HH);
    zero_f32_kernel<<<(BB * VV + 255) / 256, 256, 0, stream>>>(copybuf, BB * VV);
    copy_f32_kernel<<<(BB * HH + 255) / 256, 256, 0, stream>>>(h0s, h0in,            BB * HH);
    copy_f32_kernel<<<(BB * HH + 255) / 256, 256, 0, stream>>>(h1s, h0in + BB * HH,  BB * HH);
    copy_f32_kernel<<<(BB * HH + 255) / 256, 256, 0, stream>>>(c0s, c0in,            BB * HH);
    copy_f32_kernel<<<(BB * HH + 255) / 256, 256, 0, stream>>>(c1s, c0in + BB * HH,  BB * HH);

    // ---- sequential decode over T timesteps ----
    for (int t = 0; t < TT; ++t) {
        // Stage A0=[emb|feed|h0_prev], A1 tail=h1_prev
        prep_inputs_kernel<<<(BB * K0 + BB * HH + 255) / 256, 256, 0, stream>>>(
            tokens, t, E, feed, h0s, h1s, A0, A1);

        // LSTM layer 0: writes h0s, c0s, and bf16 h0_new into A1[:, 0:1024]
        lstm_gate_kernel<K0><<<16, 128, 0, stream>>>(
            A0, W0_bf, bsum0, c0s, h0s, nullptr, A1, K1);

        // LSTM layer 1: writes h1s, c1s, hbuf[t], bf16 htop into A2[:, 0:1024]
        lstm_gate_kernel<K1><<<16, 128, 0, stream>>>(
            A1, W1_bf, bsum1, c1s, h1s, hbuf + (size_t)t * BB * HH, A2, K1);

        // keyv[t] = htop @ Wk.T + bk  (incremental — rows t'<t already done)
        gemm32x4_kernel<HH, K1><<<4, 128, 0, stream>>>(
            A2, Wk_bf, bk, keyv + (size_t)t * BB * HH, HH, nullptr, 0, HH);

        // attention: softmax over t'<=t, copy scatter, summary -> A2[:,1024:]
        attention_kernel<<<BB, 256, 0, stream>>>(
            keyv, hbuf, h1s, tokens, t, copybuf, A2);

        // comb = [htop|summary] @ Wc.T + bc -> feed (fp32) + combbf (bf16)
        gemm32x4_kernel<K1, K1><<<4, 128, 0, stream>>>(
            A2, Wc_bf, bc, feed, HH, combbf, HH, HH);

        // vocab projection: logits = comb @ Wp.T + bp  (32 x 32000, K=1024)
        gemm32x4_kernel<HH, HH><<<125, 128, 0, stream>>>(
            combbf, Wp_bf, bp, logits, VV, nullptr, 0, VV);

        // softmax stats + copy-weight, then final log-mixture output
        rowstats_kernel<<<BB, 256, 0, stream>>>(logits, maxv, sumv, cwv);
        output_kernel<<<(BB * VV + 255) / 256, 256, 0, stream>>>(
            logits, maxv, sumv, cwv, copybuf, out, t);

        // restore copybuf to zeros (touched entries only)
        copyclear_kernel<<<t + 1, BB, 0, stream>>>(tokens, copybuf);
    }
}
